// Attention_67430986547910
// MI455X (gfx1250) — compile-verified, matrix operand order checked by reading, not run on hardware
//
#include <hip/hip_runtime.h>

typedef __attribute__((ext_vector_type(16))) _Float16 v16h;
typedef __attribute__((ext_vector_type(8)))  float    v8f;
typedef __attribute__((ext_vector_type(4)))  unsigned int v4u;
typedef __attribute__((ext_vector_type(8)))  int      v8i;
typedef __attribute__((ext_vector_type(4)))  int      v4i;

#define BATCH   4
#define SEQ     2048
#define DMODEL  512
#define HEADS   8
#define DH      64
#define QK_SCALE 0.125f      // 64^-0.5

#if defined(__gfx1250__) && __has_builtin(__builtin_amdgcn_tensor_load_to_lds) && \
    __has_builtin(__builtin_amdgcn_s_wait_tensorcnt)
#define USE_TDM 1
#endif

// ---------------------------------------------------------------------------
// WMMA fragment loaders (wave32, v_wmma_f32_16x16x32_f16)
// A-matrix 16x32 (MxK), row-major source, ld in elements (ISA 7.12.2):
//   lanes 0-15: M=lane, VGPR j<4 -> K=2j,2j+1 ; VGPR j>=4 -> K=16+2(j-4),..
//   lanes 16-31: K offsets shifted by +8.
// Also used (row<->col) to fetch B = K^T fragments from row-major K tiles.
// ---------------------------------------------------------------------------
__device__ __forceinline__ v16h load_a16x32(const _Float16* __restrict__ src,
                                            int ld, int lane) {
  const int row = lane & 15;
  const int g   = (lane >> 4) & 1;
  const _Float16* p = src + row * ld;
  v16h a;
#pragma unroll
  for (int v = 0; v < 8; ++v) {
    const int k0 = (v < 4) ? (g * 8 + 2 * v) : (16 + g * 8 + 2 * (v - 4));
    a[2 * v]     = p[k0];
    a[2 * v + 1] = p[k0 + 1];
  }
  return a;
}

// B-matrix 32x16 (KxN), source row-major [K][...], ld in elements.
__device__ __forceinline__ v16h load_b32x16(const _Float16* __restrict__ src,
                                            int ld, int lane) {
  const int col = lane & 15;
  const int g   = (lane >> 4) & 1;
  v16h b;
#pragma unroll
  for (int v = 0; v < 8; ++v) {
    const int k0 = (v < 4) ? (g * 8 + 2 * v) : (16 + g * 8 + 2 * (v - 4));
    b[2 * v]     = src[k0 * ld + col];
    b[2 * v + 1] = src[(k0 + 1) * ld + col];
  }
  return b;
}

// ---------------------------------------------------------------------------
// Kernel 1: qkv = x @ w_qkv   (f32 in, f16 out, head-major [B,H,S,Dh])
// 128 threads = 4 waves, 64x64 tile per block, K blocked by 32 through LDS.
// ---------------------------------------------------------------------------
#define XS_LD 40   // 32 + 8 halfs pad (80B rows, 16B aligned)
#define WS_LD 72   // 64 + 8 halfs pad (144B rows, 16B aligned)

__global__ __launch_bounds__(128) void qkv_gemm_f16(
    const float* __restrict__ x, const float* __restrict__ w,
    _Float16* __restrict__ qh, _Float16* __restrict__ kh,
    _Float16* __restrict__ vh) {
  __shared__ __align__(16) _Float16 xs[64 * XS_LD];
  __shared__ __align__(16) _Float16 wsh[32 * WS_LD];

  const int tid  = threadIdx.x;
  const int lane = tid & 31;
  const int wave = tid >> 5;
  const int hi   = (lane >> 4) & 1;
  const int cl   = lane & 15;
  const int m0   = blockIdx.x * 64;   // row block in [B*S, D] view
  const int n0   = blockIdx.y * 64;   // col block in [., 3D]

  v8f acc[4];
#pragma unroll
  for (int n = 0; n < 4; ++n) acc[n] = (v8f){};

  for (int kb = 0; kb < DMODEL; kb += 32) {
    // stage x tile 64x32 (f32 -> f16)
    for (int i = tid; i < (64 * 32) / 4; i += 128) {
      const int r = i >> 3;
      const int c = (i & 7) * 4;
      const float4 f = *(const float4*)(x + (size_t)(m0 + r) * DMODEL + kb + c);
      _Float16* d = xs + r * XS_LD + c;
      d[0] = (_Float16)f.x; d[1] = (_Float16)f.y;
      d[2] = (_Float16)f.z; d[3] = (_Float16)f.w;
    }
    // stage w tile 32x64 (f32 -> f16)
    for (int i = tid; i < (32 * 64) / 4; i += 128) {
      const int r = i >> 4;
      const int c = (i & 15) * 4;
      const float4 f = *(const float4*)(w + (size_t)(kb + r) * (3 * DMODEL) + n0 + c);
      _Float16* d = wsh + r * WS_LD + c;
      d[0] = (_Float16)f.x; d[1] = (_Float16)f.y;
      d[2] = (_Float16)f.z; d[3] = (_Float16)f.w;
    }
    __syncthreads();

    const v16h a = load_a16x32(xs + (wave * 16) * XS_LD, XS_LD, lane);
#pragma unroll
    for (int n = 0; n < 4; ++n) {
      const v16h b = load_b32x16(wsh + n * 16, WS_LD, lane);
      acc[n] = __builtin_amdgcn_wmma_f32_16x16x32_f16(
          false, a, false, b, (short)0, acc[n], false, false);
    }
    __syncthreads();
  }

  // scatter to head-major f16 workspace; fold softmax scale into Q
#pragma unroll
  for (int n = 0; n < 4; ++n) {
    const int gcol  = n0 + n * 16 + cl;
    const int which = gcol / DMODEL;       // 0=q 1=k 2=v (uniform per block)
    const int dcol  = gcol % DMODEL;
    const int h     = dcol >> 6;
    const int dh    = dcol & 63;
    _Float16* dst = (which == 0) ? qh : (which == 1) ? kh : vh;
    const float s = (which == 0) ? QK_SCALE : 1.0f;
#pragma unroll
    for (int r = 0; r < 8; ++r) {
      const int grow = m0 + wave * 16 + hi * 8 + r;
      const int b    = grow >> 11;         // / SEQ
      const int ss   = grow & (SEQ - 1);
      dst[((((size_t)b * HEADS + h) * SEQ) + ss) * DH + dh] =
          (_Float16)(acc[n][r] * s);
    }
  }
}

// ---------------------------------------------------------------------------
// TDM tile load: 32x64 f16 tile from a row-major [SEQ, DH] tensor into LDS,
// with hardware row padding 64 halfs data + 8 halfs pad (KV_LD = 72).
// D# per CDNA5 ISA 8.3/8.4: group0 {count, lds_addr, global_addr, type=2},
// group1 {data_size=2B, pad 32dw interval / 4dw amount, dims, strides}.
// ---------------------------------------------------------------------------
#define KV_LD 72   // 64+8 halfs pad (144B rows)
#define P_LD  40   // 32+8 halfs pad

#ifdef USE_TDM
__device__ __forceinline__ void tdm_load_tile32x64(unsigned lds_off,
                                                   const _Float16* gsrc) {
  const unsigned long long ga = (unsigned long long)(size_t)gsrc;
  v4u g0;
  g0[0] = 1u;                                          // count=1, user-mode
  g0[1] = lds_off;                                     // lds_addr (bytes)
  g0[2] = (unsigned)(ga & 0xFFFFFFFFu);                // global_addr[31:0]
  g0[3] = (unsigned)((ga >> 32) & 0x01FFFFFFu) | (2u << 30);  // [56:32], type=2
  v8i g1;
  g1[0] = (int)((1u << 16)      // data_size = 2 bytes
              | (1u << 20)      // pad_enable
              | (4u << 22)      // pad_interval: 32 DWORDs (=64 halfs)
              | (3u << 25));    // pad_amount: 4 DWORDs (=8 halfs)
  g1[1] = (int)((unsigned)DH << 16);        // tensor_dim0 lo16 (=64)
  g1[2] = (int)((unsigned)SEQ << 16);       // dim0 hi | tensor_dim1 lo16
  g1[3] = (int)((unsigned)DH << 16);        // dim1 hi | tile_dim0 = 64
  g1[4] = 32;                               // tile_dim1 = 32, tile_dim2 = 0
  g1[5] = DH;                               // tensor_dim0_stride = 64
  g1[6] = 0;
  g1[7] = 0;
  const v4i z = {0, 0, 0, 0};
#if __clang_major__ >= 23
  const v8i z8 = {0, 0, 0, 0, 0, 0, 0, 0};
  __builtin_amdgcn_tensor_load_to_lds(g0, g1, z, z, z8, 0);
#else
  __builtin_amdgcn_tensor_load_to_lds(g0, g1, z, z, 0);
#endif
}
#endif

// ---------------------------------------------------------------------------
// Kernel 2: flash attention, double-buffered KV pipeline.
// Block = (b*H+h, 64-query tile), 4 waves; each wave holds 16 query rows in
// registers and streams 32-key KV tiles. Tile t+1 is fetched (TDM async copy
// when available) while tile t is consumed by WMMA + online softmax.
// ---------------------------------------------------------------------------
__global__ __launch_bounds__(128) void flash_attn_f16(
    const _Float16* __restrict__ qg, const _Float16* __restrict__ kg,
    const _Float16* __restrict__ vg, float* __restrict__ out) {
  __shared__ __align__(16) _Float16 kt[2][32 * KV_LD];
  __shared__ __align__(16) _Float16 vt[2][32 * KV_LD];
  __shared__ __align__(16) _Float16 pt[4 * 16 * P_LD];   // per-wave P staging

  const int tid  = threadIdx.x;
  const int lane = tid & 31;
  const int wave = tid >> 5;
  const int hi   = (lane >> 4) & 1;
  const int cl   = lane & 15;
  const int bh   = blockIdx.y;             // b*H + h
  const int q0   = blockIdx.x * 64;

  const _Float16* qbase = qg + (size_t)bh * SEQ * DH;
  const _Float16* kbase = kg + (size_t)bh * SEQ * DH;
  const _Float16* vbase = vg + (size_t)bh * SEQ * DH;

  // resident Q fragments: dh 0..31 and 32..63
  const v16h qa0 = load_a16x32(qbase + (size_t)(q0 + wave * 16) * DH, DH, lane);
  const v16h qa1 = load_a16x32(qbase + (size_t)(q0 + wave * 16) * DH + 32, DH, lane);

  float m_i[8], l_i[8];
  v8f acc_o[4];
#pragma unroll
  for (int r = 0; r < 8; ++r) { m_i[r] = -3.0e38f; l_i[r] = 0.0f; }
#pragma unroll
  for (int n = 0; n < 4; ++n) acc_o[n] = (v8f){};

  _Float16* pw = pt + wave * 16 * P_LD;
  const int NT = SEQ / 32;

  // ---- prologue: stage tile 0 into buffer 0 ----
#ifdef USE_TDM
  if (wave == 0) {
    tdm_load_tile32x64((unsigned)(size_t)&kt[0][0], kbase);
    tdm_load_tile32x64((unsigned)(size_t)&vt[0][0], vbase);
  }
#else
  for (int i = tid; i < (32 * DH) / 8; i += 128) {
    const int r = i >> 3;
    const int c = (i & 7) * 8;
    *(uint4*)(&kt[0][0] + r * KV_LD + c) = *(const uint4*)(kbase + (size_t)r * DH + c);
    *(uint4*)(&vt[0][0] + r * KV_LD + c) = *(const uint4*)(vbase + (size_t)r * DH + c);
  }
#endif

  for (int t = 0; t < NT; ++t) {
    const int cur = t & 1;
#ifdef USE_TDM
    if (wave == 0) __builtin_amdgcn_s_wait_tensorcnt(0);
#endif
    __syncthreads();   // tile t resident; all waves done with buffer cur^1

    // ---- issue fetch of tile t+1 into the other buffer ----
    if (t + 1 < NT) {
#ifdef USE_TDM
      if (wave == 0) {
        tdm_load_tile32x64((unsigned)(size_t)&kt[cur ^ 1][0],
                           kbase + (size_t)(t + 1) * 32 * DH);
        tdm_load_tile32x64((unsigned)(size_t)&vt[cur ^ 1][0],
                           vbase + (size_t)(t + 1) * 32 * DH);
      }
#else
      for (int i = tid; i < (32 * DH) / 8; i += 128) {
        const int r = i >> 3;
        const int c = (i & 7) * 8;
        *(uint4*)(&kt[cur ^ 1][0] + r * KV_LD + c) =
            *(const uint4*)(kbase + (size_t)((t + 1) * 32 + r) * DH + c);
        *(uint4*)(&vt[cur ^ 1][0] + r * KV_LD + c) =
            *(const uint4*)(vbase + (size_t)((t + 1) * 32 + r) * DH + c);
      }
#endif
    }

    const _Float16* kbuf = &kt[cur][0];
    const _Float16* vbuf = &vt[cur][0];

    // ---- S = Q K^T : two 16-key subtiles, K blocked 32+32 over Dh ----
    v8f s0 = (v8f){}, s1 = (v8f){};
    {
      const v16h b00 = load_a16x32(kbuf,                   KV_LD, lane);
      const v16h b01 = load_a16x32(kbuf + 32,              KV_LD, lane);
      const v16h b10 = load_a16x32(kbuf + 16 * KV_LD,      KV_LD, lane);
      const v16h b11 = load_a16x32(kbuf + 16 * KV_LD + 32, KV_LD, lane);
      s0 = __builtin_amdgcn_wmma_f32_16x16x32_f16(false, qa0, false, b00, (short)0, s0, false, false);
      s0 = __builtin_amdgcn_wmma_f32_16x16x32_f16(false, qa1, false, b01, (short)0, s0, false, false);
      s1 = __builtin_amdgcn_wmma_f32_16x16x32_f16(false, qa0, false, b10, (short)0, s1, false, false);
      s1 = __builtin_amdgcn_wmma_f32_16x16x32_f16(false, qa1, false, b11, (short)0, s1, false, false);
    }

    // ---- online softmax (scale folded into Q); 16-lane row reductions ----
#pragma unroll
    for (int r = 0; r < 8; ++r) {
      const float v0 = s0[r], v1 = s1[r];
      float mx = fmaxf(v0, v1);
#pragma unroll
      for (int off = 8; off >= 1; off >>= 1) mx = fmaxf(mx, __shfl_xor(mx, off, 32));
      const float mnew = fmaxf(m_i[r], mx);
      const float p0 = __expf(v0 - mnew);
      const float p1 = __expf(v1 - mnew);
      float rs = p0 + p1;
#pragma unroll
      for (int off = 8; off >= 1; off >>= 1) rs += __shfl_xor(rs, off, 32);
      const float corr = __expf(m_i[r] - mnew);
      l_i[r] = l_i[r] * corr + rs;
      m_i[r] = mnew;
#pragma unroll
      for (int n = 0; n < 4; ++n) acc_o[n][r] *= corr;
      const int prow = hi * 8 + r;                // C-layout -> LDS row-major
      pw[prow * P_LD + cl]      = (_Float16)p0;
      pw[prow * P_LD + 16 + cl] = (_Float16)p1;
    }

    // ---- O += P V  (A = P 16x32, B = V 32x16 per dh subtile) ----
    const v16h pa = load_a16x32(pw, P_LD, lane);
#pragma unroll
    for (int n = 0; n < 4; ++n) {
      const v16h vb = load_b32x16(vbuf + n * 16, KV_LD, lane);
      acc_o[n] = __builtin_amdgcn_wmma_f32_16x16x32_f16(
          false, pa, false, vb, (short)0, acc_o[n], false, false);
    }
  }

  // ---- epilogue: O / l, back to [B, S, D] f32 ----
  const int h = bh & (HEADS - 1);
  const int b = bh >> 3;
#pragma unroll
  for (int r = 0; r < 8; ++r) {
    const float inv_l = 1.0f / l_i[r];
    const int grow = q0 + wave * 16 + hi * 8 + r;
#pragma unroll
    for (int n = 0; n < 4; ++n) {
      out[(((size_t)b * SEQ + grow) * DMODEL) + h * DH + n * 16 + cl] =
          acc_o[n][r] * inv_l;
    }
  }
}

// ---------------------------------------------------------------------------
extern "C" void kernel_launch(void* const* d_in, const int* in_sizes, int n_in,
                              void* d_out, int out_size, void* d_ws, size_t ws_size,
                              hipStream_t stream) {
  (void)in_sizes; (void)n_in; (void)out_size; (void)ws_size;
  const float* x = (const float*)d_in[0];       // [4, 2048, 512]
  const float* w = (const float*)d_in[1];       // [512, 1536]
  float* out = (float*)d_out;                   // [4, 2048, 512]

  const size_t per = (size_t)BATCH * HEADS * SEQ * DH;  // 4,194,304 halfs
  _Float16* qh = (_Float16*)d_ws;
  _Float16* kh = qh + per;
  _Float16* vh = kh + per;                      // total 24 MB of d_ws

  qkv_gemm_f16<<<dim3((BATCH * SEQ) / 64, (3 * DMODEL) / 64), 128, 0, stream>>>(
      x, w, qh, kh, vh);
  flash_attn_f16<<<dim3(SEQ / 64, BATCH * HEADS), 128, 0, stream>>>(
      qh, kh, vh, out);
}
